// Model_37014028156995
// MI455X (gfx1250) — compile-verified, matrix operand order checked by reading, not run on hardware
//
#include <hip/hip_runtime.h>
#include <hip/hip_bf16.h>
#include <math.h>

typedef __attribute__((ext_vector_type(16))) __bf16 v16bf;
typedef __attribute__((ext_vector_type(8)))  float  v8f;

#define EPSF 1e-5f

#define NROWS   512           // B*S
#define RG_ROWS 64            // rows per MLR block (4 wmma row-tiles)
#define N_RG    (NROWS / RG_ROWS)      // 8 row groups
#define N_VC    25            // V chunks
#define VC_TILES 80           // 16-col tiles per chunk (25*80*16 = 32000)

// ---------------- block-wide sum over 256 threads (8 wave32) ----------------
__device__ __forceinline__ float bsum(float v, float* red) {
  #pragma unroll
  for (int off = 16; off > 0; off >>= 1) v += __shfl_xor(v, off, 32);
  const int wid = threadIdx.x >> 5;
  if ((threadIdx.x & 31) == 0) red[wid] = v;
  __syncthreads();
  if (threadIdx.x == 0) {
    float s = 0.f;
    #pragma unroll
    for (int w = 0; w < 8; ++w) s += red[w];
    red[8] = s;
  }
  __syncthreads();
  float s = red[8];
  __syncthreads();
  return s;
}

// ---------------- hyperbolic primitives: thread j owns element j ----------------
__device__ __forceinline__ float d_mob_add(float x, float y, float* red) {
  float xy = bsum(x * y, red);
  float x2 = bsum(x * x, red);
  float y2 = bsum(y * y, red);
  float num = (1.f + 2.f * xy + y2) * x + (1.f - x2) * y;
  float den = 1.f + 2.f * xy + x2 * y2;
  return num / fmaxf(den, EPSF);
}
__device__ __forceinline__ float d_exp0(float v, float* red) {
  float n  = sqrtf(bsum(v * v, red));
  float ns = fmaxf(n, 1e-8f);
  return tanhf(n) * v / ns;
}
__device__ __forceinline__ float d_log0(float x, float* red) {
  float n  = sqrtf(bsum(x * x, red));
  float ns = fmaxf(n, 1e-8f);
  float c  = fminf(fmaxf(ns, 1e-8f), 1.f - EPSF);
  return atanhf(c) * x / ns;
}
__device__ __forceinline__ float d_proj(float x, float* red) {
  float n  = sqrtf(bsum(x * x, red));
  float ns = fmaxf(n, 1e-8f);
  return (n > 1.f - EPSF) ? x * (1.f - EPSF) / ns : x;
}
// Mobius matrix-vector: proj(exp0(M @ log0(x))), M is [256,256] row-major
__device__ float d_mv(const float* __restrict__ M, float x, float* red, float* vec) {
  float u = d_log0(x, red);
  vec[threadIdx.x] = u;
  __syncthreads();
  float acc = 0.f;
  const float* row = M + (size_t)threadIdx.x * 256;
  #pragma unroll 8
  for (int i = 0; i < 256; ++i) acc += row[i] * vec[i];
  __syncthreads();
  return d_proj(d_exp0(acc, red), red);
}

// ---------------- GRU scan: one block per batch row, 256 threads ----------------
__global__ void gru_kernel(const int* __restrict__ ids, const float* __restrict__ emb,
                           const float* Wr, const float* Ur, const float* br,
                           const float* Wz, const float* Uz, const float* bz,
                           const float* Wh, const float* Uh, const float* bh,
                           float* __restrict__ feats) {
  __shared__ float red[16];
  __shared__ float vec[256];
  const int b = blockIdx.x;
  const int j = threadIdx.x;
  float h = 0.f;
  for (int s = 0; s < 128; ++s) {
    const int tok = ids[b * 128 + s];
    const float x = emb[(size_t)tok * 256 + j];

    float mh = d_mv(Wr, h, red, vec);
    float mx = d_mv(Ur, x, red, vec);
    float pr = d_mob_add(d_mob_add(mh, mx, red), br[j], red);
    float r  = 1.f / (1.f + __expf(-d_log0(pr, red)));

    mh = d_mv(Wz, h, red, vec);
    mx = d_mv(Uz, x, red, vec);
    float pz = d_mob_add(d_mob_add(mh, mx, red), bz[j], red);
    float z  = 1.f / (1.f + __expf(-d_log0(pz, red)));

    float rh = d_proj(d_exp0(r * d_log0(h, red), red), red);

    mh = d_mv(Wh, rh, red, vec);
    mx = d_mv(Uh, x, red, vec);
    float ph   = d_mob_add(d_mob_add(mh, mx, red), bh[j], red);
    float htil = d_proj(d_exp0(tanhf(d_log0(ph, red)), red), red);

    float delta = d_exp0(z * d_log0(d_mob_add(-h, htil, red), red), red);
    h = d_proj(d_mob_add(h, delta, red), red);

    feats[((size_t)b * 128 + s) * 256 + j] = h;   // [B,S,H] = reshape order
  }
}

// ---------------- prep: feats -> bf16 + row norms ----------------
__global__ void prep_feats(const float* __restrict__ feats, __bf16* __restrict__ fb,
                           float* __restrict__ x2) {
  const int n = blockIdx.x * blockDim.x + threadIdx.x;
  if (n >= NROWS) return;
  const float* src = feats + (size_t)n * 256;
  __bf16* dst = fb + (size_t)n * 256;
  float s = 0.f;
  for (int i = 0; i < 256; ++i) {
    float v = src[i];
    dst[i] = (__bf16)v;
    s += v * v;
  }
  x2[n] = s;
}

// ---------------- prep: p/a -> bf16 + per-vocab stats ----------------
__global__ void prep_vocab(const float* __restrict__ p, const float* __restrict__ a,
                           __bf16* __restrict__ pb, __bf16* __restrict__ ab,
                           float* __restrict__ p2, float* __restrict__ pa,
                           float* __restrict__ an) {
  const int v = blockIdx.x * blockDim.x + threadIdx.x;
  if (v >= 32000) return;
  const float* pr = p + (size_t)v * 256;
  const float* ar = a + (size_t)v * 256;
  __bf16* pbo = pb + (size_t)v * 256;
  __bf16* abo = ab + (size_t)v * 256;
  float sp2 = 0.f, spa = 0.f, sa2 = 0.f;
  for (int i = 0; i < 256; ++i) {
    float pv = pr[i], av = ar[i];
    pbo[i] = (__bf16)pv;
    abo[i] = (__bf16)av;
    sp2 += pv * pv; spa += pv * av; sa2 += av * av;
  }
  p2[v] = sp2; pa[v] = spa; an[v] = sqrtf(sa2);
}

// ---------------- MLR head: WMMA GEMM + fused online logsumexp ----------------
// Grid (N_RG, N_VC). Block = 64 rows (4 row-tiles, A staged in LDS, shared by
// all 8 waves and reused 4x per B fragment); waves split the chunk's 80 V-tiles.
// Emits per-(chunk,row) partial (max, sumexp) for a later merge.
__global__ __launch_bounds__(256, 1) void mlr_kernel(
    const __bf16* __restrict__ fb,
    const __bf16* __restrict__ pb,
    const __bf16* __restrict__ ab,
    const float* __restrict__ x2,
    const float* __restrict__ p2,
    const float* __restrict__ pa,
    const float* __restrict__ an,
    float* __restrict__ pm,         // [N_VC][NROWS] partial max
    float* __restrict__ ps) {       // [N_VC][NROWS] partial sumexp
  __shared__ __align__(16) __bf16 atile[RG_ROWS * 256];  // 32 KB A tile
  __shared__ float sx2[RG_ROWS];
  __shared__ float smax[8][RG_ROWS];
  __shared__ float ssum[8][RG_ROWS];
  const int tid  = threadIdx.x;
  const int wid  = tid >> 5;
  const int lane = tid & 31;
  const int half = lane >> 4;      // K-half of the 16-bit fragment layout
  const int nlo  = lane & 15;      // M (for A) / N (for B) index
  const int rowbase = blockIdx.x * RG_ROWS;
  const int vc      = blockIdx.y;
  const int koff = half * 8;

  // Cooperative stage of the 64x256 bf16 A tile into LDS (2048 x 16B).
  {
    const uint4* gsrc = (const uint4*)(fb + (size_t)rowbase * 256);
    uint4* ldst = (uint4*)atile;
    #pragma unroll
    for (int i = 0; i < 8; ++i) ldst[tid + i * 256] = gsrc[tid + i * 256];
  }
  if (tid < RG_ROWS) sx2[tid] = x2[rowbase + tid];
  __syncthreads();

  float mrow[4][8], srow[4][8];
  #pragma unroll
  for (int rt = 0; rt < 4; ++rt)
    #pragma unroll
    for (int r = 0; r < 8; ++r) { mrow[rt][r] = -3.0e38f; srow[rt][r] = 0.f; }

  for (int t = wid; t < VC_TILES; t += 8) {
    const int col = vc * (VC_TILES * 16) + t * 16 + nlo;
    v8f accp[4], acca[4];
    #pragma unroll
    for (int rt = 0; rt < 4; ++rt) {
      accp[rt] = (v8f){0.f,0.f,0.f,0.f,0.f,0.f,0.f,0.f};
      acca[rt] = (v8f){0.f,0.f,0.f,0.f,0.f,0.f,0.f,0.f};
    }
    const __bf16* prow = pb + (size_t)col * 256 + koff;
    const __bf16* arow = ab + (size_t)col * 256 + koff;
    #pragma unroll
    for (int kc = 0; kc < 8; ++kc) {
      v16bf bp, ba;
      const __bf16* sp = prow + kc * 32;
      const __bf16* sa = arow + kc * 32;
      #pragma unroll
      for (int i = 0; i < 8; ++i) {
        bp[i] = sp[i];  bp[8 + i] = sp[16 + i];
        ba[i] = sa[i];  ba[8 + i] = sa[16 + i];
      }
      #pragma unroll
      for (int rt = 0; rt < 4; ++rt) {
        v16bf af;   // A fragment from LDS, dead after its 2 WMMAs
        const __bf16* asrc = atile + (rt * 16 + nlo) * 256 + kc * 32 + koff;
        #pragma unroll
        for (int i = 0; i < 8; ++i) {
          af[i] = asrc[i];  af[8 + i] = asrc[16 + i];
        }
        accp[rt] = __builtin_amdgcn_wmma_f32_16x16x32_bf16(false, af, false, bp,
                                                 (short)0, accp[rt], false, false);
        acca[rt] = __builtin_amdgcn_wmma_f32_16x16x32_bf16(false, af, false, ba,
                                                 (short)0, acca[rt], false, false);
      }
    }
    const float p2c = p2[col];
    const float pac = pa[col];
    const float anc = an[col];
    const float lam = 2.f / fmaxf(1.f - p2c, EPSF);
    #pragma unroll
    for (int rt = 0; rt < 4; ++rt) {
      #pragma unroll
      for (int r = 0; r < 8; ++r) {
        const float xp  = accp[rt][r];
        const float xa  = acca[rt][r];
        const float x2r = sx2[rt * 16 + half * 8 + r];
        const float Dv    = fmaxf(1.f - 2.f * xp + p2c * x2r, EPSF);
        const float alpha = (1.f - 2.f * xp + x2r) / Dv;
        const float beta  = (1.f - p2c) / Dv;
        const float za    = -alpha * pac + beta * xa;
        const float z2    = alpha * alpha * p2c - 2.f * alpha * beta * xp
                          + beta * beta * x2r;
        const float logit = lam * anc *
            asinhf(2.f * za / (fmaxf(1.f - z2, EPSF) * fmaxf(anc, 1e-8f)));
        const float nm = fmaxf(mrow[rt][r], logit);
        srow[rt][r] = srow[rt][r] * __expf(mrow[rt][r] - nm) + __expf(logit - nm);
        mrow[rt][r] = nm;
      }
    }
  }

  // merge across the 16 lanes of each half (xor bits 0..3 stay in-half)
  #pragma unroll
  for (int off = 1; off < 16; off <<= 1) {
    #pragma unroll
    for (int rt = 0; rt < 4; ++rt) {
      #pragma unroll
      for (int r = 0; r < 8; ++r) {
        const float om = __shfl_xor(mrow[rt][r], off, 32);
        const float os = __shfl_xor(srow[rt][r], off, 32);
        const float nm = fmaxf(mrow[rt][r], om);
        srow[rt][r] = srow[rt][r] * __expf(mrow[rt][r] - nm) + os * __expf(om - nm);
        mrow[rt][r] = nm;
      }
    }
  }
  if (nlo == 0) {
    #pragma unroll
    for (int rt = 0; rt < 4; ++rt)
      #pragma unroll
      for (int r = 0; r < 8; ++r) {
        smax[wid][rt * 16 + half * 8 + r] = mrow[rt][r];
        ssum[wid][rt * 16 + half * 8 + r] = srow[rt][r];
      }
  }
  __syncthreads();
  if (tid < RG_ROWS) {
    float m = -3.0e38f, s = 0.f;
    #pragma unroll
    for (int w = 0; w < 8; ++w) {
      const float om = smax[w][tid], os = ssum[w][tid];
      const float nm = fmaxf(m, om);
      s = s * __expf(m - nm) + os * __expf(om - nm);
      m = nm;
    }
    pm[(size_t)vc * NROWS + rowbase + tid] = m;
    ps[(size_t)vc * NROWS + rowbase + tid] = s;
  }
}

// ---------------- merge per-chunk partials into per-row logsumexp ----------------
__global__ void lse_merge(const float* __restrict__ pm, const float* __restrict__ ps,
                          float* __restrict__ lse) {
  const int row = blockIdx.x * blockDim.x + threadIdx.x;
  if (row >= NROWS) return;
  float m = -3.0e38f, s = 0.f;
  for (int vc = 0; vc < N_VC; ++vc) {
    const float om = pm[(size_t)vc * NROWS + row];
    const float os = ps[(size_t)vc * NROWS + row];
    const float nm = fmaxf(m, om);
    s = s * __expf(m - nm) + os * __expf(om - nm);
    m = nm;
  }
  lse[row] = m + __logf(s);
}

// ---------------- fp32 target logits (one wave per row) ----------------
__global__ void tgt_logit_kernel(const float* __restrict__ feats,
                                 const float* __restrict__ x2,
                                 const float* __restrict__ p,
                                 const float* __restrict__ a,
                                 const float* __restrict__ p2,
                                 const float* __restrict__ pa,
                                 const float* __restrict__ an,
                                 const int* __restrict__ tgt,
                                 float* __restrict__ tl) {
  const int wid  = threadIdx.x >> 5;
  const int lane = threadIdx.x & 31;
  const int row  = blockIdx.x * 8 + wid;
  if (row >= NROWS) return;
  const int t = tgt[row];
  const float* f  = feats + (size_t)row * 256;
  const float* pr = p + (size_t)t * 256;
  const float* ar = a + (size_t)t * 256;
  float xp = 0.f, xa = 0.f;
  #pragma unroll 2
  for (int i = lane; i < 256; i += 32) {
    const float fv = f[i];
    xp += fv * pr[i];
    xa += fv * ar[i];
  }
  #pragma unroll
  for (int off = 16; off > 0; off >>= 1) {
    xp += __shfl_xor(xp, off, 32);
    xa += __shfl_xor(xa, off, 32);
  }
  if (lane == 0) {
    const float x2r = x2[row];
    const float p2c = p2[t];
    const float pac = pa[t];
    const float anc = an[t];
    const float lam = 2.f / fmaxf(1.f - p2c, EPSF);
    const float Dv    = fmaxf(1.f - 2.f * xp + p2c * x2r, EPSF);
    const float alpha = (1.f - 2.f * xp + x2r) / Dv;
    const float beta  = (1.f - p2c) / Dv;
    const float za    = -alpha * pac + beta * xa;
    const float z2    = alpha * alpha * p2c - 2.f * alpha * beta * xp
                      + beta * beta * x2r;
    tl[row] = lam * anc *
        asinhf(2.f * za / (fmaxf(1.f - z2, EPSF) * fmaxf(anc, 1e-8f)));
  }
}

// ---------------- final loss reduction ----------------
__global__ void loss_kernel(const float* __restrict__ lse, const float* __restrict__ tl,
                            float* __restrict__ out) {
  __shared__ float red[8];
  float acc = 0.f;
  for (int r = threadIdx.x; r < NROWS; r += 256) acc += lse[r] - tl[r];
  #pragma unroll
  for (int off = 16; off > 0; off >>= 1) acc += __shfl_xor(acc, off, 32);
  if ((threadIdx.x & 31) == 0) red[threadIdx.x >> 5] = acc;
  __syncthreads();
  if (threadIdx.x == 0) {
    float s = 0.f;
    for (int w = 0; w < 8; ++w) s += red[w];
    out[0] = s / (float)NROWS;
  }
}

extern "C" void kernel_launch(void* const* d_in, const int* in_sizes, int n_in,
                              void* d_out, int out_size, void* d_ws, size_t ws_size,
                              hipStream_t stream) {
  const int*   ids   = (const int*)d_in[0];
  const int*   tgt   = (const int*)d_in[1];
  const float* emb   = (const float*)d_in[2];
  const float* Wr    = (const float*)d_in[3];
  const float* Ur    = (const float*)d_in[4];
  const float* br    = (const float*)d_in[5];
  const float* Wz    = (const float*)d_in[6];
  const float* Uz    = (const float*)d_in[7];
  const float* bz    = (const float*)d_in[8];
  const float* Wh    = (const float*)d_in[9];
  const float* Uh    = (const float*)d_in[10];
  const float* bh    = (const float*)d_in[11];
  const float* p_mlr = (const float*)d_in[12];
  const float* a_mlr = (const float*)d_in[13];
  float* out = (float*)d_out;

  char* ws = (char*)d_ws;
  size_t off = 0;
  auto carve = [&](size_t bytes) -> void* {
    void* p = ws + off;
    off += (bytes + 255) & ~(size_t)255;
    return p;
  };
  float*  feats = (float*)carve((size_t)NROWS * 256 * 4);
  __bf16* fb    = (__bf16*)carve((size_t)NROWS * 256 * 2);
  float*  x2    = (float*)carve(NROWS * 4);
  __bf16* pb    = (__bf16*)carve((size_t)32000 * 256 * 2);
  __bf16* ab    = (__bf16*)carve((size_t)32000 * 256 * 2);
  float*  p2    = (float*)carve(32000 * 4);
  float*  pa    = (float*)carve(32000 * 4);
  float*  an    = (float*)carve(32000 * 4);
  float*  pm    = (float*)carve((size_t)N_VC * NROWS * 4);
  float*  ps    = (float*)carve((size_t)N_VC * NROWS * 4);
  float*  lse   = (float*)carve(NROWS * 4);
  float*  tl    = (float*)carve(NROWS * 4);
  (void)in_sizes; (void)n_in; (void)out_size; (void)ws_size;

  prep_vocab<<<dim3(125), dim3(256), 0, stream>>>(p_mlr, a_mlr, pb, ab, p2, pa, an);
  gru_kernel<<<dim3(4), dim3(256), 0, stream>>>(ids, emb, Wr, Ur, br, Wz, Uz, bz,
                                                Wh, Uh, bh, feats);
  prep_feats<<<dim3(2), dim3(256), 0, stream>>>(feats, fb, x2);
  mlr_kernel<<<dim3(N_RG, N_VC), dim3(256), 0, stream>>>(fb, pb, ab, x2, p2, pa, an,
                                                         pm, ps);
  tgt_logit_kernel<<<dim3(64), dim3(256), 0, stream>>>(feats, x2, p_mlr, a_mlr,
                                                       p2, pa, an, tgt, tl);
  lse_merge<<<dim3(2), dim3(256), 0, stream>>>(pm, ps, lse);
  loss_kernel<<<dim3(1), dim3(256), 0, stream>>>(lse, tl, out);
}
